// MonomialGATNet_41120016892604
// MI455X (gfx1250) — matrix operand order repaired and checked by reading, not compiled, not run on hardware
//
#include <hip/hip_runtime.h>
#include <math.h>

typedef __attribute__((ext_vector_type(16))) _Float16 v16h;
typedef __attribute__((ext_vector_type(8)))  float    v8f;

#define N_NODES   100000
#define E_EDGES   1600000
#define E_TOT     (E_EDGES + N_NODES)   // with self-loops
#define F_IN      512
#define H1_HEADS  8
#define D1_DIM    8
#define F_MID     64                    // H1*D1
#define C_CLS     16
#define NEG_SLOPE 0.2f

// ---------------------------------------------------------------------------
// B pre-pack: convert fp32 B[K x NC] into per-lane WMMA f16 fragments.
// Fragment (kstep ks, n-tile t, lane) holds 16 halves:
//   j=0..15 -> B[(ks*32 + (lane<16?0:8) + j + (j<8?0:8))*NC + t*16 + (lane&15)]
// stored contiguously (32B per lane) so the GEMM can load it with b128s.
// ---------------------------------------------------------------------------
template <int KDIM, int NC>
__global__ void pack_b_kernel(const float* __restrict__ B, _Float16* __restrict__ Bp) {
    constexpr int NT    = NC / 16;
    constexpr int NFRAG = (KDIM / 32) * NT * 32;
    int tid = blockIdx.x * blockDim.x + threadIdx.x;
    if (tid >= NFRAG) return;
    int lane = tid & 31;
    int t    = (tid >> 5) % NT;
    int ks   = tid / (32 * NT);
    int krow = ks * 32 + ((lane < 16) ? 0 : 8);
    int col  = t * 16 + (lane & 15);
    _Float16* dst = Bp + (size_t)tid * 16;
#pragma unroll
    for (int j = 0; j < 16; ++j)
        dst[j] = (_Float16)B[(size_t)(krow + j + (j < 8 ? 0 : 8)) * NC + col];
}

// ---------------------------------------------------------------------------
// WMMA GEMM:  C[M x NC] = A[M x K] * B[K x NC]
//  - A fp32 in memory, converted to f16 in registers (streams from HBM once)
//  - B pre-packed f16 fragments, staged in LDS, loaded via ds_load_b128
//  - f32 accumulation via v_wmma_f32_16x16x32_f16
// One wave computes a 16 x NC tile; M must be a multiple of 16.
// ---------------------------------------------------------------------------
template <int KDIM, int NC>
__global__ __launch_bounds__(128)
void gemm_wmma_kernel(const float* __restrict__ A, const _Float16* __restrict__ Bp,
                      float* __restrict__ C, int M) {
    constexpr int NT = NC / 16;
    constexpr int BH = (KDIM / 32) * NT * 32 * 16;  // halves in packed B
    __shared__ _Float16 bsh[BH];                    // 64KB for GEMM1, 2KB for GEMM2

    // cooperative stage of packed B into LDS (16B per thread-iteration)
    {
        const uint4* src = (const uint4*)Bp;
        uint4*       dst = (uint4*)bsh;
        for (int i = threadIdx.x; i < BH / 8; i += blockDim.x) dst[i] = src[i];
    }
    __syncthreads();

    const int wave = blockIdx.x * (blockDim.x >> 5) + (threadIdx.x >> 5);
    const int lane = threadIdx.x & 31;
    const int m0 = wave * 16;
    if (m0 >= M) return;                    // wave-uniform exit (EXEC stays all-1s)

    const int arow  = m0 + (lane & 15);     // A: lanes 0-15 -> M=0..15, 16-31 -> M=0..15
    const int aKoff = (lane < 16) ? 0 : 8;  // A comps 0-7 = K+off..+7, 8-15 = K+off+16..+23

    v8f acc[NT];
    v8f zacc = {};
#pragma unroll
    for (int t = 0; t < NT; ++t) acc[t] = zacc;

    for (int k0 = 0; k0 < KDIM; k0 += 32) {
        const float* ap = A + (size_t)arow * KDIM + k0 + aKoff;
        __builtin_prefetch(ap + 64, 0, 1);  // global_prefetch_b8
        v16h afrag;
#pragma unroll
        for (int j = 0; j < 8; ++j) {
            afrag[j]     = (_Float16)ap[j];
            afrag[j + 8] = (_Float16)ap[j + 16];
        }
        const int ks = k0 >> 5;
#pragma unroll
        for (int t = 0; t < NT; ++t) {
            v16h bfrag = *(const v16h*)&bsh[(size_t)(((ks * NT + t) * 32) + lane) * 16];
            acc[t] = __builtin_amdgcn_wmma_f32_16x16x32_f16(
                false, afrag, false, bfrag, (short)0, acc[t], false, false);
        }
    }

    // C/D layout: VGPR v, lanes 0-15 -> M = v, lanes 16-31 -> M = v + 8
    const int rbase = m0 + ((lane < 16) ? 0 : 8);
    const int ncol  = lane & 15;
#pragma unroll
    for (int t = 0; t < NT; ++t) {
#pragma unroll
        for (int v = 0; v < 8; ++v) {
            C[(size_t)(rbase + v) * NC + t * 16 + ncol] = acc[t][v];
        }
    }
}

// ---------------------------------------------------------------------------
// Helpers / elementwise kernels
// ---------------------------------------------------------------------------
__global__ void fill_kernel(float* __restrict__ p, float v, int n) {
    int i = blockIdx.x * blockDim.x + threadIdx.x;
    if (i < n) p[i] = v;
}

__global__ void elu_inplace_kernel(float* __restrict__ p, int n) {
    int i = blockIdx.x * blockDim.x + threadIdx.x;
    if (i < n) {
        float x = p[i];
        p[i] = (x > 0.0f) ? x : expm1f(x);
    }
}

// per-node attention terms: es[n,h] = sum_d feat[n,h,d]*a_s[h,d] (same for ed)
template <int H, int D>
__global__ void attn_terms_kernel(const float* __restrict__ feat,
                                  const float* __restrict__ a_s,
                                  const float* __restrict__ a_d,
                                  float* __restrict__ es, float* __restrict__ ed,
                                  int n) {
    int i = blockIdx.x * blockDim.x + threadIdx.x;
    if (i >= n) return;
    const float* f = feat + (size_t)i * H * D;
#pragma unroll
    for (int h = 0; h < H; ++h) {
        float ss = 0.0f, sd = 0.0f;
#pragma unroll
        for (int d = 0; d < D; ++d) {
            float v = f[h * D + d];
            ss += v * a_s[h * D + d];
            sd += v * a_d[h * D + d];
        }
        es[i * H + h] = ss;
        ed[i * H + h] = sd;
    }
}

__device__ __forceinline__ void edge_ends(const int* __restrict__ ei, int e,
                                          int& s, int& d) {
    if (e < E_EDGES) { s = ei[e]; d = ei[E_EDGES + e]; }
    else             { s = e - E_EDGES; d = s; }       // self-loop
}

__device__ __forceinline__ float leaky(float x) {
    return (x > 0.0f) ? x : NEG_SLOPE * x;
}

// sign-aware float atomic max (memory initialized to -inf)
__device__ __forceinline__ void atomicMaxFloat(float* addr, float val) {
    if (val >= 0.0f) atomicMax((int*)addr, __float_as_int(val));
    else             atomicMin((unsigned int*)addr, __float_as_uint(val));
}

template <int H>
__global__ void edge_max_kernel(const int* __restrict__ ei,
                                const float* __restrict__ es,
                                const float* __restrict__ ed,
                                float* __restrict__ m) {
    int e = blockIdx.x * blockDim.x + threadIdx.x;
    if (e >= E_TOT) return;
    int s, d; edge_ends(ei, e, s, d);
#pragma unroll
    for (int h = 0; h < H; ++h) {
        float sc = leaky(es[(size_t)s * H + h] + ed[(size_t)d * H + h]);
        atomicMaxFloat(&m[(size_t)d * H + h], sc);
    }
}

template <int H>
__global__ void edge_sum_kernel(const int* __restrict__ ei,
                                const float* __restrict__ es,
                                const float* __restrict__ ed,
                                const float* __restrict__ m,
                                float* __restrict__ ssum) {
    int e = blockIdx.x * blockDim.x + threadIdx.x;
    if (e >= E_TOT) return;
    int s, d; edge_ends(ei, e, s, d);
#pragma unroll
    for (int h = 0; h < H; ++h) {
        float sc = leaky(es[(size_t)s * H + h] + ed[(size_t)d * H + h]);
        atomicAdd(&ssum[(size_t)d * H + h], expf(sc - m[(size_t)d * H + h]));
    }
}

// one thread per (edge, feature-channel); alpha recomputed per channel
template <int H, int D>
__global__ void edge_agg_kernel(const int* __restrict__ ei,
                                const float* __restrict__ es,
                                const float* __restrict__ ed,
                                const float* __restrict__ m,
                                const float* __restrict__ ssum,
                                const float* __restrict__ feat,
                                float* __restrict__ agg) {
    constexpr int FD = H * D;
    long long t = (long long)blockIdx.x * blockDim.x + threadIdx.x;
    if (t >= (long long)E_TOT * FD) return;
    int e = (int)(t / FD);
    int c = (int)(t % FD);
    int h = c / D;
    int s, d; edge_ends(ei, e, s, d);
    float sc = leaky(es[(size_t)s * H + h] + ed[(size_t)d * H + h]);
    float alpha = expf(sc - m[(size_t)d * H + h]) / ssum[(size_t)d * H + h];
    atomicAdd(&agg[(size_t)d * FD + c], alpha * feat[(size_t)s * FD + c]);
}

__global__ void logsoftmax16_kernel(const float* __restrict__ h2,
                                    float* __restrict__ out, int n) {
    int i = blockIdx.x * blockDim.x + threadIdx.x;
    if (i >= n) return;
    const float* row = h2 + (size_t)i * C_CLS;
    float mx = -__builtin_inff();
#pragma unroll
    for (int c = 0; c < C_CLS; ++c) mx = fmaxf(mx, row[c]);
    float ssum = 0.0f;
#pragma unroll
    for (int c = 0; c < C_CLS; ++c) ssum += expf(row[c] - mx);
    float lse = mx + logf(ssum);
#pragma unroll
    for (int c = 0; c < C_CLS; ++c) out[(size_t)i * C_CLS + c] = row[c] - lse;
}

// ---------------------------------------------------------------------------
// Launcher
// ---------------------------------------------------------------------------
static inline int ceil_div(long long a, int b) { return (int)((a + b - 1) / b); }

extern "C" void kernel_launch(void* const* d_in, const int* in_sizes, int n_in,
                              void* d_out, int out_size, void* d_ws, size_t ws_size,
                              hipStream_t stream) {
    const float* x   = (const float*)d_in[0];
    const int*   ei  = (const int*)  d_in[1];   // [2, E] flat: src then dst
    const float* W1  = (const float*)d_in[2];
    const float* a1s = (const float*)d_in[3];
    const float* a1d = (const float*)d_in[4];
    const float* W2  = (const float*)d_in[5];
    const float* a2s = (const float*)d_in[6];
    const float* a2d = (const float*)d_in[7];
    float* out = (float*)d_out;

    // workspace carve-up (~79 MB + packed B)
    float* ws    = (float*)d_ws;
    float* h1pre = ws;  ws += (size_t)N_NODES * F_MID;   // x@W1, pre-aggregation feats
    float* es1   = ws;  ws += (size_t)N_NODES * H1_HEADS;
    float* ed1   = ws;  ws += (size_t)N_NODES * H1_HEADS;
    float* m1    = ws;  ws += (size_t)N_NODES * H1_HEADS;
    float* s1    = ws;  ws += (size_t)N_NODES * H1_HEADS;
    float* agg1  = ws;  ws += (size_t)N_NODES * F_MID;   // aggregated, then ELU in place
    float* h2pre = ws;  ws += (size_t)N_NODES * C_CLS;
    float* es2   = ws;  ws += (size_t)N_NODES;
    float* ed2   = ws;  ws += (size_t)N_NODES;
    float* m2    = ws;  ws += (size_t)N_NODES;
    float* s2    = ws;  ws += (size_t)N_NODES;
    float* agg2  = ws;  ws += (size_t)N_NODES * C_CLS;
    _Float16* bp1 = (_Float16*)ws;                       // packed W1: 32768 halves (64KB)
    _Float16* bp2 = bp1 + (size_t)(F_IN / 32) * (F_MID / 16) * 32 * 16;
                                                         // packed W2: 1024 halves (2KB)
    const int TB = 256;
    const float NEG_INF = -__builtin_inff();

    // ---- pack weights into WMMA fragment layout (tiny, one-shot) ----
    pack_b_kernel<F_IN, F_MID><<<ceil_div((F_IN / 32) * (F_MID / 16) * 32, TB), TB, 0, stream>>>(W1, bp1);
    pack_b_kernel<F_MID, C_CLS><<<1, 64, 0, stream>>>(W2, bp2);

    // ---- Layer 1 ----
    {   // GEMM1: h1pre = x @ W1   (100000x512 * 512x64)
        int waves  = N_NODES / 16;            // 6250
        int blocks = ceil_div(waves, 4);      // 4 waves / 128-thread block
        gemm_wmma_kernel<F_IN, F_MID><<<blocks, 128, 0, stream>>>(x, bp1, h1pre, N_NODES);
    }
    attn_terms_kernel<H1_HEADS, D1_DIM><<<ceil_div(N_NODES, TB), TB, 0, stream>>>(
        h1pre, a1s, a1d, es1, ed1, N_NODES);

    fill_kernel<<<ceil_div((long long)N_NODES * H1_HEADS, TB), TB, 0, stream>>>(
        m1, NEG_INF, N_NODES * H1_HEADS);
    fill_kernel<<<ceil_div((long long)N_NODES * H1_HEADS, TB), TB, 0, stream>>>(
        s1, 0.0f, N_NODES * H1_HEADS);
    fill_kernel<<<ceil_div((long long)N_NODES * F_MID, TB), TB, 0, stream>>>(
        agg1, 0.0f, N_NODES * F_MID);

    edge_max_kernel<H1_HEADS><<<ceil_div(E_TOT, TB), TB, 0, stream>>>(ei, es1, ed1, m1);
    edge_sum_kernel<H1_HEADS><<<ceil_div(E_TOT, TB), TB, 0, stream>>>(ei, es1, ed1, m1, s1);
    edge_agg_kernel<H1_HEADS, D1_DIM>
        <<<ceil_div((long long)E_TOT * F_MID, TB), TB, 0, stream>>>(
            ei, es1, ed1, m1, s1, h1pre, agg1);

    elu_inplace_kernel<<<ceil_div((long long)N_NODES * F_MID, TB), TB, 0, stream>>>(
        agg1, N_NODES * F_MID);

    // ---- Layer 2 ----
    {   // GEMM2: h2pre = elu(agg1) @ W2   (100000x64 * 64x16)
        int waves  = N_NODES / 16;
        int blocks = ceil_div(waves, 4);
        gemm_wmma_kernel<F_MID, C_CLS><<<blocks, 128, 0, stream>>>(agg1, bp2, h2pre, N_NODES);
    }
    attn_terms_kernel<1, C_CLS><<<ceil_div(N_NODES, TB), TB, 0, stream>>>(
        h2pre, a2s, a2d, es2, ed2, N_NODES);

    fill_kernel<<<ceil_div(N_NODES, TB), TB, 0, stream>>>(m2, NEG_INF, N_NODES);
    fill_kernel<<<ceil_div(N_NODES, TB), TB, 0, stream>>>(s2, 0.0f, N_NODES);
    fill_kernel<<<ceil_div((long long)N_NODES * C_CLS, TB), TB, 0, stream>>>(
        agg2, 0.0f, N_NODES * C_CLS);

    edge_max_kernel<1><<<ceil_div(E_TOT, TB), TB, 0, stream>>>(ei, es2, ed2, m2);
    edge_sum_kernel<1><<<ceil_div(E_TOT, TB), TB, 0, stream>>>(ei, es2, ed2, m2, s2);
    edge_agg_kernel<1, C_CLS>
        <<<ceil_div((long long)E_TOT * C_CLS, TB), TB, 0, stream>>>(
            ei, es2, ed2, m2, s2, h2pre, agg2);

    logsoftmax16_kernel<<<ceil_div(N_NODES, TB), TB, 0, stream>>>(agg2, out, N_NODES);
}
